// TransformerEncoder_54331336294756
// MI455X (gfx1250) — matrix operand order repaired and checked by reading, not compiled, not run on hardware
//
#include <hip/hip_runtime.h>

// ---------------------------------------------------------------------------
// Types for CDNA5 WMMA (wave32): bf16 A/B fragments, f32 accumulator
// ---------------------------------------------------------------------------
typedef __attribute__((ext_vector_type(16))) __bf16 v16bf;
typedef __attribute__((ext_vector_type(8)))  float  v8f;

union V16 { v16bf v; uint4 q[2]; };

__device__ __forceinline__ unsigned short f2bf(float f) {
    unsigned u = __float_as_uint(f);
    u += 0x7FFFu + ((u >> 16) & 1u);   // round-to-nearest-even
    return (unsigned short)(u >> 16);
}

// CDNA5 async global->LDS DMA (ASYNCcnt-tracked), per-lane 16B.
__device__ __forceinline__ void async_b128(unsigned lds_off, const void* gptr) {
    asm volatile("global_load_async_to_lds_b128 %0, %1, off"
                 :: "v"(lds_off), "v"((unsigned long long)(uintptr_t)gptr)
                 : "memory");
}
__device__ __forceinline__ void wait_async0() {
    asm volatile("s_wait_asynccnt 0x0" ::: "memory");
}
// Wait until <=4 async ops outstanding (this wave): the 4 newest (one tile)
// may remain in flight; all older tiles are guaranteed resident.
__device__ __forceinline__ void wait_async4() {
    asm volatile("s_wait_asynccnt 0x4" ::: "memory");
}

// ---------------------------------------------------------------------------
// Elementwise f32 -> bf16 cast (4 elems / thread)
// ---------------------------------------------------------------------------
__global__ __launch_bounds__(256)
void cast_f32_bf16(const float* __restrict__ in, unsigned short* __restrict__ out,
                   size_t n4) {
    size_t idx = (size_t)blockIdx.x * 256 + threadIdx.x;
    if (idx >= n4) return;
    float4 v = *(const float4*)(in + idx * 4);
    union { uint2 u; unsigned short h[4]; } t;
    t.h[0] = f2bf(v.x); t.h[1] = f2bf(v.y); t.h[2] = f2bf(v.z); t.h[3] = f2bf(v.w);
    *(uint2*)(out + idx * 4) = t.u;
}

// ---------------------------------------------------------------------------
// Transpose-cast: in f32 [R,C] row-major  ->  out bf16 [C,R] row-major
// 32x32 tiles via LDS; grid (C/32, R/32), block (32,8)
// ---------------------------------------------------------------------------
__global__ __launch_bounds__(256)
void cast_transpose_f32_bf16(const float* __restrict__ in,
                             unsigned short* __restrict__ out, int R, int C) {
    __shared__ float t[32][33];
    const int bx = blockIdx.x * 32;   // C offset
    const int by = blockIdx.y * 32;   // R offset
    const int tx = threadIdx.x, ty = threadIdx.y;
#pragma unroll
    for (int j = 0; j < 4; ++j)
        t[ty + j * 8][tx] = in[(size_t)(by + ty + j * 8) * C + (bx + tx)];
    __syncthreads();
#pragma unroll
    for (int j = 0; j < 4; ++j)
        out[(size_t)(bx + ty + j * 8) * R + (by + tx)] = f2bf(t[tx][ty + j * 8]);
}

// ---------------------------------------------------------------------------
// bf16 WMMA GEMM: C[M,N] = act(A[M,K] @ B^T + bias + addsrc)
//   A: [M,K] row-major bf16;  B: [N,K] row-major bf16 (ldb = row stride)
// Block tile 128x128, BK=32, 256 threads = 8 waves (2m x 4n); each wave owns a
// 64x32 sub-tile = 4x2 WMMA accumulators.
// Data movement: 3-stage GLOBAL_LOAD_ASYNC_TO_LDS_B128 pipeline; DMA issued
// two tiles ahead; steady-state loop is unconditional (issue / compute /
// s_wait_asynccnt 4 / barrier); last two k-steps are peeled. Needs K/32 >= 3.
// Outputs: Cf (f32) and/or Cb (bf16); cbT=1 writes Cb transposed [N,M].
// ---------------------------------------------------------------------------
__global__ __launch_bounds__(256)
void gemm_bf16_wmma(const unsigned short* __restrict__ A,
                    const unsigned short* __restrict__ B,
                    const float* __restrict__ bias,
                    const float* __restrict__ addsrc,
                    float* __restrict__ Cf,
                    unsigned short* __restrict__ Cb,
                    int M, int N, int K, int ldb, int cbT, int act) {
    __shared__ __align__(16) unsigned short sA[3][128 * 40];
    __shared__ __align__(16) unsigned short sB[3][128 * 40];

    const int tid  = threadIdx.x;
    const int lane = tid & 31;
    const int wave = tid >> 5;
    const int m0 = blockIdx.y * 128;
    const int n0 = blockIdx.x * 128;
    const int wm = (wave >> 2) * 64;   // wave row offset in block tile
    const int wn = (wave & 3) * 32;    // wave col offset in block tile
    const int fr  = lane & 15;
    const int akb = (lane < 16) ? 0 : 8;    // A fragment K base (ISA layout)
    const int bkb = (lane < 16) ? 0 : 8;    // B fragment K base (ISA layout)

    v8f acc[4][2] = {};

    // global->LDS tiling: each thread DMAs 2x16B for A and 2x16B for B
    const int row  = tid >> 2;           // 0..63 (+64 for 2nd chunk)
    const int koff = (tid & 3) * 8;      // halfword offset within 32-wide K

    unsigned baseA[3], baseB[3];
#pragma unroll
    for (int b = 0; b < 3; ++b) {
        baseA[b] = (unsigned)(uintptr_t)&sA[b][0];
        baseB[b] = (unsigned)(uintptr_t)&sB[b][0];
    }
    const unsigned offL0 = (unsigned)((row * 40 + koff) * 2);
    const unsigned offL1 = (unsigned)(((row + 64) * 40 + koff) * 2);

    auto issue = [&](int k0, int b) {
        async_b128(baseA[b] + offL0, A + (size_t)(m0 + row)      * K + (k0 + koff));
        async_b128(baseA[b] + offL1, A + (size_t)(m0 + row + 64) * K + (k0 + koff));
        async_b128(baseB[b] + offL0, B + (size_t)(n0 + row)      * ldb + (k0 + koff));
        async_b128(baseB[b] + offL1, B + (size_t)(n0 + row + 64) * ldb + (k0 + koff));
    };

    auto compute = [&](int b) {
        const unsigned short* pa = &sA[b][0];
        const unsigned short* pb = &sB[b][0];
        V16 fa[4], fb[2];
#pragma unroll
        for (int i = 0; i < 4; ++i) {
            const unsigned short* p = pa + (wm + i * 16 + fr) * 40 + akb;
            fa[i].q[0] = *(const uint4*)p;          // K = akb .. akb+7
            fa[i].q[1] = *(const uint4*)(p + 16);   // K = 16+akb .. 16+akb+7
        }
#pragma unroll
        for (int j = 0; j < 2; ++j) {
            const unsigned short* p = pb + (wn + j * 16 + fr) * 40 + bkb;
            fb[j].q[0] = *(const uint4*)p;          // K = bkb .. bkb+7
            fb[j].q[1] = *(const uint4*)(p + 16);   // K = 16+bkb .. 16+bkb+7
        }
#pragma unroll
        for (int i = 0; i < 4; ++i)
#pragma unroll
            for (int j = 0; j < 2; ++j)
                acc[i][j] = __builtin_amdgcn_wmma_f32_16x16x32_bf16(
                    false, fa[i].v, false, fb[j].v, (short)0, acc[i][j],
                    false, false);
    };

    const int nt = K >> 5;          // number of 32-wide K tiles (>= 3 here)
    issue(0, 0);
    issue(32, 1);
    wait_async4();                  // tile 0 resident; tile 1 may be in flight
    __syncthreads();

    int bc = 0;                     // buffer being computed
    int bi = 2;                     // buffer being filled
    for (int t = 0; t < nt - 2; ++t) {
        issue((t + 2) << 5, bi);    // DMA two tiles ahead
        compute(bc);
        wait_async4();              // tile t+1 resident across the barrier
        __syncthreads();
        bc = (bc == 2) ? 0 : bc + 1;
        bi = (bi == 2) ? 0 : bi + 1;
    }
    // peeled tail: tiles nt-2 and nt-1 (no further DMA)
    compute(bc);
    bc = (bc == 2) ? 0 : bc + 1;
    wait_async0();                  // drain final tile
    __syncthreads();
    compute(bc);

    // ---- epilogue: VGPR e of a 16x16 f32 tile = row (e | e+8), col = lane&15
#pragma unroll
    for (int i = 0; i < 4; ++i) {
        const int rbase = m0 + wm + i * 16 + ((lane < 16) ? 0 : 8);
#pragma unroll
        for (int j = 0; j < 2; ++j) {
            const int col = n0 + wn + j * 16 + fr;
            const float bvs = bias ? bias[col] : 0.0f;
#pragma unroll
            for (int e = 0; e < 8; ++e) {
                const int r = rbase + e;
                float v = acc[i][j][e] + bvs;
                if (addsrc) v += addsrc[(size_t)r * N + col];
                if (act) v = tanhf(v);
                if (Cf) Cf[(size_t)r * N + col] = v;
                if (Cb) {
                    if (cbT) Cb[(size_t)col * M + r] = f2bf(v);
                    else     Cb[(size_t)r * N + col] = f2bf(v);
                }
            }
        }
    }
}

// ---------------------------------------------------------------------------
// Row softmax with mask applied BEFORE softmax: p = softmax(mask * s / 32)
// One 256-thread block per row of [rows, cols], cols multiple of 1024.
// ---------------------------------------------------------------------------
__global__ __launch_bounds__(256)
void softmax_mask(const float* __restrict__ scores,
                  const float* __restrict__ mask,
                  unsigned short* __restrict__ prob, int cols) {
    const int row = blockIdx.x;
    const int tid = threadIdx.x;
    const float* srow = scores + (size_t)row * cols;
    const float* grow = mask   + (size_t)row * cols;
    const int nit = cols >> 10;       // iterations of 1024 cols
    float vals[16];
    float mx = -3.0e38f;
    for (int it = 0; it < nit; ++it) {
        int c = it * 1024 + tid * 4;
        float4 s = *(const float4*)(srow + c);
        float4 g = *(const float4*)(grow + c);
        float v0 = g.x * s.x * 0.03125f;   // 1/sqrt(1024)
        float v1 = g.y * s.y * 0.03125f;
        float v2 = g.z * s.z * 0.03125f;
        float v3 = g.w * s.w * 0.03125f;
        vals[it * 4 + 0] = v0; vals[it * 4 + 1] = v1;
        vals[it * 4 + 2] = v2; vals[it * 4 + 3] = v3;
        mx = fmaxf(mx, fmaxf(fmaxf(v0, v1), fmaxf(v2, v3)));
    }
#pragma unroll
    for (int off = 16; off > 0; off >>= 1)
        mx = fmaxf(mx, __shfl_xor(mx, off, 32));
    __shared__ float redm[8];
    __shared__ float reds[8];
    if ((tid & 31) == 0) redm[tid >> 5] = mx;
    __syncthreads();
    float bm = redm[0];
#pragma unroll
    for (int wv = 1; wv < 8; ++wv) bm = fmaxf(bm, redm[wv]);

    float sum = 0.0f;
    for (int i = 0; i < nit * 4; ++i) {
        vals[i] = __expf(vals[i] - bm);
        sum += vals[i];
    }
#pragma unroll
    for (int off = 16; off > 0; off >>= 1)
        sum += __shfl_xor(sum, off, 32);
    if ((tid & 31) == 0) reds[tid >> 5] = sum;
    __syncthreads();
    float bs = 0.0f;
#pragma unroll
    for (int wv = 0; wv < 8; ++wv) bs += reds[wv];
    const float inv = 1.0f / bs;

    for (int it = 0; it < nit; ++it) {
        int c = it * 1024 + tid * 4;
        union { uint2 u; unsigned short h[4]; } t;
        t.h[0] = f2bf(vals[it * 4 + 0] * inv);
        t.h[1] = f2bf(vals[it * 4 + 1] * inv);
        t.h[2] = f2bf(vals[it * 4 + 2] * inv);
        t.h[3] = f2bf(vals[it * 4 + 3] * inv);
        *(uint2*)(prob + (size_t)row * cols + c) = t.u;
    }
}

// ---------------------------------------------------------------------------
// BatchNorm column stats (biased variance) over [rows, cols]
// grid = cols/32 blocks; 256 threads = 8 row-groups x 32 cols
// ---------------------------------------------------------------------------
__global__ __launch_bounds__(256)
void bn_stats(const float* __restrict__ x, float* __restrict__ mean,
              float* __restrict__ rstd, int rows, int cols) {
    const int tid = threadIdx.x;
    const int c  = blockIdx.x * 32 + (tid & 31);
    const int rg = tid >> 5;
    float s = 0.0f, s2 = 0.0f;
    for (int r = rg; r < rows; r += 8) {
        float v = x[(size_t)r * cols + c];
        s += v; s2 += v * v;
    }
    __shared__ float ls[8][33];
    __shared__ float ls2[8][33];
    ls[rg][tid & 31]  = s;
    ls2[rg][tid & 31] = s2;
    __syncthreads();
    if (rg == 0) {
        float a = 0.0f, b = 0.0f;
#pragma unroll
        for (int g = 0; g < 8; ++g) { a += ls[g][tid]; b += ls2[g][tid]; }
        float mu  = a / (float)rows;
        float var = b / (float)rows - mu * mu;
        mean[c] = mu;
        rstd[c] = rsqrtf(var + 1e-5f);
    }
}

// ---------------------------------------------------------------------------
// BatchNorm apply: out = [tanh]((x - mean) * rstd * gamma + beta)
// Optional f32 out (may be in-place) and/or bf16 out. cols is a power of 2.
// ---------------------------------------------------------------------------
__global__ __launch_bounds__(256)
void bn_apply(const float* __restrict__ x, const float* __restrict__ mean,
              const float* __restrict__ rstd, const float* __restrict__ gamma,
              const float* __restrict__ beta, float* __restrict__ outF,
              unsigned short* __restrict__ outB, int cols, int do_tanh,
              size_t total4) {
    size_t idx = (size_t)blockIdx.x * 256 + threadIdx.x;
    if (idx >= total4) return;
    size_t i = idx * 4;
    int c0 = (int)(i & (size_t)(cols - 1));
    float4 v4 = *(const float4*)(x + i);
    float vv[4] = {v4.x, v4.y, v4.z, v4.w};
    float o[4];
#pragma unroll
    for (int e = 0; e < 4; ++e) {
        int c = c0 + e;
        float t = (vv[e] - mean[c]) * rstd[c] * gamma[c] + beta[c];
        o[e] = do_tanh ? tanhf(t) : t;
    }
    if (outF) *(float4*)(outF + i) = make_float4(o[0], o[1], o[2], o[3]);
    if (outB) {
        union { uint2 u; unsigned short h[4]; } t;
        t.h[0] = f2bf(o[0]); t.h[1] = f2bf(o[1]);
        t.h[2] = f2bf(o[2]); t.h[3] = f2bf(o[3]);
        *(uint2*)(outB + i) = t.u;
    }
}

// ---------------------------------------------------------------------------
// Host orchestration
// ---------------------------------------------------------------------------
extern "C" void kernel_launch(void* const* d_in, const int* in_sizes, int n_in,
                              void* d_out, int out_size, void* d_ws, size_t ws_size,
                              hipStream_t stream) {
    (void)in_sizes; (void)n_in; (void)out_size; (void)ws_size;
    const float* src  = (const float*)d_in[0];
    const float* anc1 = (const float*)d_in[1];
    const float* anc2 = (const float*)d_in[2];
    const float* lgr  = (const float*)d_in[3];
    const float* Wq   = (const float*)d_in[4];
    const float* bq   = (const float*)d_in[5];
    const float* Wv   = (const float*)d_in[6];
    const float* bv   = (const float*)d_in[7];
    const float* W1   = (const float*)d_in[8];
    const float* b1   = (const float*)d_in[9];
    const float* W2   = (const float*)d_in[10];
    const float* b2   = (const float*)d_in[11];
    const float* g1   = (const float*)d_in[12];
    const float* be1  = (const float*)d_in[13];
    const float* g2   = (const float*)d_in[14];
    const float* be2  = (const float*)d_in[15];
    const float* Wd   = (const float*)d_in[16];
    const float* bd   = (const float*)d_in[17];
    const float* gd   = (const float*)d_in[18];
    const float* bed  = (const float*)d_in[19];

    const int N = 8192, Mnc = 4096, QD = 4096, D = 1024, FFD = 2048, VD = 512;

    char* w = (char*)d_ws;
    size_t off = 0;
    auto alloc = [&](size_t bytes) -> void* {
        void* p = w + off;
        off += (bytes + 255) & ~(size_t)255;
        return p;
    };
    unsigned short* srcb = (unsigned short*)alloc((size_t)N * QD * 2);   // reused: prob bf16
    unsigned short* a1b  = (unsigned short*)alloc((size_t)Mnc * QD * 2); // reused: h bf16
    unsigned short* a2b  = (unsigned short*)alloc((size_t)Mnc * VD * 2);
    unsigned short* WqT  = (unsigned short*)alloc((size_t)QD * D * 2);   // [D,QD]
    unsigned short* WvT  = (unsigned short*)alloc((size_t)VD * D * 2);   // [D,VD]
    unsigned short* W1T  = (unsigned short*)alloc((size_t)D * FFD * 2);  // [FFD,D]
    unsigned short* W2T  = (unsigned short*)alloc((size_t)FFD * D * 2);  // [D,FFD]
    unsigned short* WdT  = (unsigned short*)alloc((size_t)D * VD * 2);   // [VD,D]
    unsigned short* Qb   = (unsigned short*)alloc((size_t)N * D * 2);
    unsigned short* Kb   = (unsigned short*)alloc((size_t)Mnc * D * 2);
    unsigned short* Vt   = (unsigned short*)alloc((size_t)Mnc * D * 2);  // [D,Mnc]
    float* scores        = (float*)alloc((size_t)N * Mnc * 4);
    float* x             = (float*)alloc((size_t)N * D * 4);
    unsigned short* xnb  = (unsigned short*)alloc((size_t)N * D * 2);
    float* x2            = (float*)alloc((size_t)N * D * 4);
    unsigned short* x2nb = (unsigned short*)alloc((size_t)N * D * 2);
    float* y             = (float*)alloc((size_t)N * VD * 4);
    float* mean1 = (float*)alloc(D * 4);  float* rstd1 = (float*)alloc(D * 4);
    float* mean2 = (float*)alloc(D * 4);  float* rstd2 = (float*)alloc(D * 4);
    float* meand = (float*)alloc(VD * 4); float* rstdd = (float*)alloc(VD * 4);

    auto cast = [&](const float* in, unsigned short* out, size_t n) {
        size_t n4 = n / 4;
        cast_f32_bf16<<<(unsigned)((n4 + 255) / 256), 256, 0, stream>>>(in, out, n4);
    };
    auto castT = [&](const float* in, unsigned short* out, int R, int C) {
        dim3 grid(C / 32, R / 32), blk(32, 8);
        cast_transpose_f32_bf16<<<grid, blk, 0, stream>>>(in, out, R, C);
    };
    cast(src,  srcb, (size_t)N * QD);
    cast(anc1, a1b,  (size_t)Mnc * QD);
    cast(anc2, a2b,  (size_t)Mnc * VD);
    castT(Wq, WqT, QD, D);     // [QD,D] -> [D,QD]
    castT(Wv, WvT, VD, D);     // [VD,D] -> [D,VD]
    castT(W1, W1T, D, FFD);    // [D,FFD] -> [FFD,D]
    castT(W2, W2T, FFD, D);    // [FFD,D] -> [D,FFD]
    castT(Wd, WdT, D, VD);     // [D,VD] -> [VD,D]

    // gemm: C[M,N] = act(A[M,K] @ Bt[N,K]^T + bias + add); ldb = K
    auto gemm = [&](const unsigned short* A, const unsigned short* B,
                    const float* bias, const float* add, float* Cf,
                    unsigned short* Cb, int M_, int N_, int K_, int cbT_,
                    int act_) {
        dim3 grid(N_ / 128, M_ / 128);
        gemm_bf16_wmma<<<grid, 256, 0, stream>>>(A, B, bias, add, Cf, Cb,
                                                 M_, N_, K_, K_, cbT_, act_);
    };

    // Q = src@Wq+bq (bf16), K = anchor1@Wq+bq (bf16), V^T = (anchor2@Wv+bv)^T
    gemm(srcb, WqT, bq, nullptr, nullptr, Qb, N,   D, QD, 0, 0);
    gemm(a1b,  WqT, bq, nullptr, nullptr, Kb, Mnc, D, QD, 0, 0);
    gemm(a2b,  WvT, bv, nullptr, nullptr, Vt, Mnc, D, VD, 1, 0);  // -> Vt [D,Mnc]
    // scores = Q @ K^T (f32): B = Kb [Mnc, D]
    gemm(Qb, Kb, nullptr, nullptr, scores, nullptr, N, Mnc, D, 0, 0);
    // prob = softmax(mask * scores / 32) -> bf16, reusing srcb
    softmax_mask<<<N, 256, 0, stream>>>(scores, lgr, srcb, Mnc);
    // x = prob @ V (f32): B = Vt [D, Mnc]
    gemm(srcb, Vt, nullptr, nullptr, x, nullptr, N, D, Mnc, 0, 0);
    // BN1 (in-place f32 + bf16 copy)
    bn_stats<<<D / 32, 256, 0, stream>>>(x, mean1, rstd1, N, D);
    {
        size_t t4 = (size_t)N * D / 4;
        bn_apply<<<(unsigned)((t4 + 255) / 256), 256, 0, stream>>>(
            x, mean1, rstd1, g1, be1, x, xnb, D, 0, t4);
    }
    // h = tanh(xn @ W1 + b1) (bf16, reusing a1b): B = W1T [FFD, D]
    gemm(xnb, W1T, b1, nullptr, nullptr, a1b, N, FFD, D, 0, 1);
    // x2 = xn + h @ W2 + b2 (f32, residual fused): B = W2T [D, FFD]
    gemm(a1b, W2T, b2, x, x2, nullptr, N, D, FFD, 0, 0);
    // BN2
    bn_stats<<<D / 32, 256, 0, stream>>>(x2, mean2, rstd2, N, D);
    {
        size_t t4 = (size_t)N * D / 4;
        bn_apply<<<(unsigned)((t4 + 255) / 256), 256, 0, stream>>>(
            x2, mean2, rstd2, g2, be2, x2, x2nb, D, 0, t4);
    }
    // y = x2n @ Wd + bd (f32): B = WdT [VD, D]
    gemm(x2nb, WdT, bd, nullptr, y, nullptr, N, VD, D, 0, 0);
    // out = tanh(BN(y))
    bn_stats<<<VD / 32, 256, 0, stream>>>(y, meand, rstdd, N, VD);
    {
        size_t t4 = (size_t)N * VD / 4;
        bn_apply<<<(unsigned)((t4 + 255) / 256), 256, 0, stream>>>(
            y, meand, rstdd, gd, bed, (float*)d_out, nullptr, VD, 1, t4);
    }
}
